// BiLSTM_CRF_83124797047151
// MI455X (gfx1250) — matrix operand order repaired and checked by reading, not compiled, not run on hardware
//
#include <hip/hip_runtime.h>
#include <math.h>

// ---------------------------------------------------------------------------
// BiLSTM-CRF forward (V=50000, E=100, H=5, L=10, T=10, B=64, S=2048)
// Pipeline: embed-gather -> conv(WMMA f16) -> relu -> maxpool -> lin1 ->
//           10x (pregate GEMM + bidirectional recurrence) -> feats -> CRF
// ---------------------------------------------------------------------------

typedef __attribute__((ext_vector_type(16))) _Float16 v16h;
typedef __attribute__((ext_vector_type(8)))  _Float16 v8h;
typedef __attribute__((ext_vector_type(8)))  float    v8f;

#define SEQ   2048
#define BATCH 64
#define EMB   100
#define HID   5
#define NTAG  10

// ---------------- prep: embedding gather -> f16 (S+2,64,100), zero pad rows
__global__ void prep_x_kernel(const int* __restrict__ sentence,
                              const float* __restrict__ emb,
                              _Float16* __restrict__ Xg) {
  const size_t total = (size_t)(SEQ + 2) * BATCH * EMB;
  for (size_t t = (size_t)blockIdx.x * blockDim.x + threadIdx.x; t < total;
       t += (size_t)gridDim.x * blockDim.x) {
    int e = (int)(t % EMB);
    size_t r = t / EMB;
    int b = (int)(r % BATCH);
    int srow = (int)(r / BATCH);           // 0..2049, row g <-> seq pos g-1
    float v = 0.f;
    if (srow >= 1 && srow <= SEQ) {
      int tok = sentence[(size_t)b * SEQ + (srow - 1)];
      v = emb[(size_t)tok * EMB + e];
    }
    Xg[t] = (_Float16)v;
  }
}

// ---------------- prep: conv_w (o,i,3,50) f32 -> Wg (kh,i,o,64) f16, kw padded
__global__ void prep_w_kernel(const float* __restrict__ conv_w,
                              _Float16* __restrict__ Wg) {
  size_t t = (size_t)blockIdx.x * blockDim.x + threadIdx.x;  // exact 786432
  int kw = (int)(t & 63);
  int o  = (int)((t >> 6) & 63);
  int i  = (int)((t >> 12) & 63);
  int kh = (int)(t >> 18);
  float v = (kw < 50) ? conv_w[(((size_t)o * 64 + i) * 3 + kh) * 50 + kw] : 0.f;
  Wg[t] = (_Float16)v;
}

// ---------------- conv via WMMA: out[o][s][j] (64,2048,51), relu(+bias)
// block: 512 threads (16 waves), per-block s-tile of 16.
// LDS: Xs[18][64][114] f16 (262656B) + Ws[64][64] f16 (8192B)
__global__ __launch_bounds__(512) void conv_wmma_kernel(
    const _Float16* __restrict__ Xg,   // (2050,64,100)
    const _Float16* __restrict__ Wg,   // (3,64,64,64)
    const float* __restrict__ bias,    // (64)
    float* __restrict__ out)           // (64,2048,51)
{
  extern __shared__ char dynsm[];
  _Float16* Xs = (_Float16*)dynsm;              // [r 0..17][i 0..63][col 0..113]
  _Float16* Ws = Xs + 18 * 64 * 114;            // [o][kw64]
  const int s0  = blockIdx.x * 16;
  const int tid = threadIdx.x;

  // zero slab (covers col padding 100..113) via u32 stores
  for (int idx = tid; idx < 18 * 64 * 57; idx += 512)
    ((uint32_t*)Xs)[idx] = 0u;
  __syncthreads();
  // copy 18 rows x 64 ch x 100 cols (global rows s0..s0+17, pre-padded)
  for (int idx = tid; idx < 18 * 64 * 50; idx += 512) {
    int r = idx / (64 * 50); int rem = idx - r * 64 * 50;
    int i = rem / 50;        int c   = rem - i * 50;
    uint32_t v = ((const uint32_t*)(Xg + ((size_t)(s0 + r) * 64 + i) * EMB))[c];
    *(uint32_t*)&Xs[(r * 64 + i) * 114 + c * 2] = v;
  }

  const int w = tid >> 5, lane = tid & 31;
  const int ot = w & 3, o0 = ot * 16;
  const int jlo = (w >> 2) * 13;                 // j groups: 0,13,26,39
  const int n = lane & 15, h = lane >> 4;
  const int o = o0 + n;

  v8f acc[13];
#pragma unroll
  for (int jj = 0; jj < 13; ++jj)
#pragma unroll
    for (int r = 0; r < 8; ++r) acc[jj][r] = 0.f;

  __syncthreads();
  for (int kh = 0; kh < 3; ++kh) {
    for (int i = 0; i < 64; ++i) {
      const _Float16* wslice = Wg + ((size_t)(kh * 64 + i) << 12);
      __syncthreads();                 // previous readers of Ws done
      if (tid < 512) ((v8h*)Ws)[tid] = ((const v8h*)wslice)[tid];  // 8KB
      __syncthreads();
      if (kh * 64 + i < 191)
        __builtin_prefetch(wslice + 4096, 0, 1);   // -> global_prefetch_b8

      // A fragments: A[m][k] = W[o0+m][kw=k], 16x32 layout (wave32)
      const v8h* wrow = (const v8h*)(Ws + o * 64);
      v16h a0, a1;
      ((v8h*)&a0)[0] = wrow[0 + 2 * h];   // K = 16h .. 16h+7
      ((v8h*)&a0)[1] = wrow[1 + 2 * h];   // K = 16h+8 .. +7
      ((v8h*)&a1)[0] = wrow[4 + 2 * h];   // K = 32+16h
      ((v8h*)&a1)[1] = wrow[5 + 2 * h];   // K = 40+16h
      // B row base: X[s0+n+kh-1 (global) -> LDS row n+kh][i][*]
      const _Float16* xrow = Xs + ((n + kh) * 64 + i) * 114;
#pragma unroll
      for (int jj = 0; jj < 13; ++jj) {
        const int j = jlo + jj;
        v16h b0, b1;
#pragma unroll
        for (int e = 0; e < 16; ++e)
          ((_Float16*)&b0)[e] = xrow[j + 16 * h + e];       // K = 16h+e
#pragma unroll
        for (int e = 0; e < 16; ++e)
          ((_Float16*)&b1)[e] = xrow[j + 32 + 16 * h + e];  // K = 32+16h+e
        acc[jj] = __builtin_amdgcn_wmma_f32_16x16x32_f16(
            false, a0, false, b0, (short)0, acc[jj], false, false);
        acc[jj] = __builtin_amdgcn_wmma_f32_16x16x32_f16(
            false, a1, false, b1, (short)0, acc[jj], false, false);
      }
    }
  }

  // epilogue: relu(acc + bias) -> out[o][s][j]
  const int s = s0 + n;
  float bo[8];
#pragma unroll
  for (int r = 0; r < 8; ++r) bo[r] = bias[o0 + r + 8 * h];
#pragma unroll
  for (int jj = 0; jj < 13; ++jj) {
    const int j = jlo + jj;
    if (j < 51) {
#pragma unroll
      for (int r = 0; r < 8; ++r) {
        float v = acc[jj][r] + bo[r];
        v = v > 0.f ? v : 0.f;
        out[(((size_t)(o0 + r + 8 * h)) * SEQ + s) * 51 + j] = v;
      }
    }
  }
}

// ---------------- maxpool window (3s,22j) over (64,2048,51) -> (64,2048,30)
__global__ void pool_kernel(const float* __restrict__ conv,
                            float* __restrict__ pooled) {
  const size_t total = (size_t)64 * SEQ * 30;
  for (size_t t = (size_t)blockIdx.x * blockDim.x + threadIdx.x; t < total;
       t += (size_t)gridDim.x * blockDim.x) {
    int jj = (int)(t % 30);
    size_t r = t / 30;
    int s = (int)(r % SEQ);
    int o = (int)(r / SEQ);
    float mx = -3.4e38f;
    for (int ds = -1; ds <= 1; ++ds) {
      int ss = s + ds;
      if (ss < 0 || ss >= SEQ) continue;
      const float* row = conv + ((size_t)o * SEQ + ss) * 51;
      for (int j = jj; j < jj + 22; ++j) mx = fmaxf(mx, row[j]);
    }
    pooled[t] = mx;
  }
}

// ---------------- lin1: concat(embeds[b,s,100], pooled[b,s,30]) @ W.T + b
// one block per (s,b); writes x0 in (S,B,100) layout for the LSTM stack
__global__ __launch_bounds__(128) void lin1_kernel(
    const int* __restrict__ sentence, const float* __restrict__ emb,
    const float* __restrict__ pooled, const float* __restrict__ w,
    const float* __restrict__ bvec, float* __restrict__ x0) {
  const int s = blockIdx.x >> 6;
  const int b = blockIdx.x & 63;
  __shared__ float inrow[130];
  const int tid = threadIdx.x;
  if (tid < 100) {
    int tok = sentence[(size_t)b * SEQ + s];
    inrow[tid] = emb[(size_t)tok * EMB + tid];
  } else if (tid < 130) {
    inrow[tid] = pooled[((size_t)b * SEQ + s) * 30 + (tid - 100)];
  }
  __syncthreads();
  if (tid < 100) {
    const float* wr = w + tid * 130;
    float a = bvec[tid];
    for (int k = 0; k < 130; ++k) a += inrow[k] * wr[k];
    x0[((size_t)s * BATCH + b) * EMB + tid] = a;
  }
}

// ---------------- LSTM pre-gates (time-parallel part), both directions
__global__ void lstm_pregate_kernel(
    const float* __restrict__ x, int in_dim,
    const float* __restrict__ WihF, const float* __restrict__ bihF,
    const float* __restrict__ bhhF,
    const float* __restrict__ WihB, const float* __restrict__ bihB,
    const float* __restrict__ bhhB,
    float* __restrict__ preF, float* __restrict__ preB) {
  extern __shared__ char dynsm[];
  float* WF = (float*)dynsm;
  float* WB = WF + 20 * in_dim;
  float* bs = WB + 20 * in_dim;  // [40]
  for (int idx = threadIdx.x; idx < 20 * in_dim; idx += blockDim.x) {
    WF[idx] = WihF[idx]; WB[idx] = WihB[idx];
  }
  for (int idx = threadIdx.x; idx < 20; idx += blockDim.x) {
    bs[idx] = bihF[idx] + bhhF[idx];
    bs[20 + idx] = bihB[idx] + bhhB[idx];
  }
  __syncthreads();
  const size_t total = (size_t)SEQ * BATCH * 20;
  for (size_t t = (size_t)blockIdx.x * blockDim.x + threadIdx.x; t < total * 2;
       t += (size_t)gridDim.x * blockDim.x) {
    int dir = t >= total;
    size_t u = dir ? t - total : t;
    int g = (int)(u % 20);
    size_t sb = u / 20;
    const float* xr = x + sb * in_dim;
    const float* Wr = (dir ? WB : WF) + g * in_dim;
    float a = bs[dir * 20 + g];
    for (int k = 0; k < in_dim; ++k) a += xr[k] * Wr[k];
    (dir ? preB : preF)[u] = a;
  }
}

__device__ __forceinline__ float sigf(float x) { return 1.f / (1.f + expf(-x)); }

// ---------------- LSTM recurrence: grid = 2 blocks (fwd, bwd), 256 threads
__global__ __launch_bounds__(256) void lstm_scan_kernel(
    const float* __restrict__ preF, const float* __restrict__ preB,
    const float* __restrict__ WhhF, const float* __restrict__ WhhB,
    const float* __restrict__ h0, const float* __restrict__ c0,  // (2,64,5)
    float* __restrict__ xout) {                                   // (S,64,10)
  __shared__ float hbuf[BATCH * HID], cbuf[BATCH * HID];
  __shared__ float gbuf[BATCH * 20], whh[20 * HID];
  const int dir = blockIdx.x;
  const float* pre = dir ? preB : preF;
  const float* Whh = dir ? WhhB : WhhF;
  const int tid = threadIdx.x;
  for (int idx = tid; idx < 100; idx += 256) whh[idx] = Whh[idx];
  for (int idx = tid; idx < 320; idx += 256) {
    hbuf[idx] = h0[dir * 320 + idx];
    cbuf[idx] = c0[dir * 320 + idx];
  }
  __syncthreads();
  const int b = tid >> 2, q = tid & 3;
  float wr[25];
#pragma unroll
  for (int j = 0; j < 5; ++j)
#pragma unroll
    for (int k = 0; k < 5; ++k) wr[j * 5 + k] = whh[(q * 5 + j) * 5 + k];
  for (int step = 0; step < SEQ; ++step) {
    const int s = dir ? (SEQ - 1 - step) : step;
    const float* pr = pre + ((size_t)s * BATCH + b) * 20 + q * 5;
    float hreg[5];
#pragma unroll
    for (int k = 0; k < 5; ++k) hreg[k] = hbuf[b * 5 + k];
#pragma unroll
    for (int j = 0; j < 5; ++j) {
      float a = pr[j];
#pragma unroll
      for (int k = 0; k < 5; ++k) a += hreg[k] * wr[j * 5 + k];
      gbuf[b * 20 + q * 5 + j] = a;
    }
    __syncthreads();
    if (q == 0) {
#pragma unroll
      for (int j = 0; j < 5; ++j) {
        float ig = gbuf[b * 20 + j],      fg = gbuf[b * 20 + 5 + j];
        float gg = gbuf[b * 20 + 10 + j], og = gbuf[b * 20 + 15 + j];
        float c = sigf(fg) * cbuf[b * 5 + j] + sigf(ig) * tanhf(gg);
        float hh = sigf(og) * tanhf(c);
        cbuf[b * 5 + j] = c;
        hbuf[b * 5 + j] = hh;
        xout[((size_t)s * BATCH + b) * 10 + dir * 5 + j] = hh;
      }
    }
    __syncthreads();
  }
}

// ---------------- feats: (S,B,10) @ tag_w.T + tag_b -> feats (B,S,10)
__global__ void feats_kernel(const float* __restrict__ xfin,
                             const float* __restrict__ tag_w,
                             const float* __restrict__ tag_b,
                             float* __restrict__ feats) {
  const size_t total = (size_t)BATCH * SEQ * NTAG;
  for (size_t t = (size_t)blockIdx.x * blockDim.x + threadIdx.x; t < total;
       t += (size_t)gridDim.x * blockDim.x) {
    int tg = (int)(t % NTAG);
    size_t bs = t / NTAG;
    int s = (int)(bs % SEQ);
    int b = (int)(bs / SEQ);
    const float* xr = xfin + ((size_t)s * BATCH + b) * 10;
    float a = tag_b[tg];
    for (int k = 0; k < 10; ++k) a += xr[k] * tag_w[tg * 10 + k];
    feats[((size_t)b * SEQ + s) * NTAG + tg] = a;
  }
}

// ---------------- CRF: single block, 640 threads (20 wave32), alpha in LDS
__global__ __launch_bounds__(640) void crf_kernel(
    const float* __restrict__ feats, const int* __restrict__ target,
    const unsigned char* __restrict__ mask,  // numpy bool: 1 byte/elem
    const float* __restrict__ crf_start, const float* __restrict__ crf_end,
    const float* __restrict__ crf_trans, float* __restrict__ out) {
  __shared__ float tr[100], alpha[BATCH * NTAG], st[NTAG], en[NTAG];
  __shared__ float numpart[BATCH * NTAG], bsum[BATCH];
  __shared__ int cntpart[BATCH * NTAG];
  const int tid = threadIdx.x;
  const int b = tid / NTAG, t = tid % NTAG;
  if (tid < 100) tr[tid] = crf_trans[tid];
  if (tid < NTAG) { st[tid] = crf_start[tid]; en[tid] = crf_end[tid]; }
  __syncthreads();
  alpha[b * NTAG + t] = st[t] + feats[(size_t)b * SEQ * NTAG + t];
  __syncthreads();
  for (int s = 1; s < SEQ; ++s) {
    float arow[NTAG];
#pragma unroll
    for (int k = 0; k < NTAG; ++k) arow[k] = alpha[b * NTAG + k];
    float mx = -3.4e38f;
#pragma unroll
    for (int k = 0; k < NTAG; ++k) {
      float v = arow[k] + tr[k * NTAG + t];
      mx = v > mx ? v : mx;
    }
    float sum = 0.f;
#pragma unroll
    for (int k = 0; k < NTAG; ++k) sum += expf(arow[k] + tr[k * NTAG + t] - mx);
    float nxt = mx + logf(sum) + feats[((size_t)b * SEQ + s) * NTAG + t];
    float newv = (mask[(size_t)b * SEQ + s] != 0) ? nxt : arow[t];
    __syncthreads();
    alpha[b * NTAG + t] = newv;
    __syncthreads();
  }
  // numerator partials (strided over s by tag-lane)
  float np = 0.f;
  int cnt = 0;
  for (int s = t; s < SEQ; s += NTAG) {
    int m = mask[(size_t)b * SEQ + s] != 0;
    cnt += m;
    if (s >= 1 && m) {
      int tg = target[(size_t)b * SEQ + s];
      int tgp = target[(size_t)b * SEQ + s - 1];
      np += feats[((size_t)b * SEQ + s) * NTAG + tg] + tr[tgp * NTAG + tg];
    }
  }
  if (t == 0) {
    int tg0 = target[(size_t)b * SEQ];
    np += feats[(size_t)b * SEQ * NTAG + tg0] + st[tg0];
  }
  numpart[tid] = np;
  cntpart[tid] = cnt;
  __syncthreads();
  if (t == 0) {
    float num = 0.f; int c = 0;
    for (int k = 0; k < NTAG; ++k) { num += numpart[b * NTAG + k]; c += cntpart[b * NTAG + k]; }
    num += en[target[(size_t)b * SEQ + (c - 1)]];
    float mx = -3.4e38f;
    for (int k = 0; k < NTAG; ++k) {
      float v = alpha[b * NTAG + k] + en[k];
      mx = v > mx ? v : mx;
    }
    float sum = 0.f;
    for (int k = 0; k < NTAG; ++k) sum += expf(alpha[b * NTAG + k] + en[k] - mx);
    bsum[b] = num - (mx + logf(sum));
  }
  __syncthreads();
  if (tid == 0) {
    float L = 0.f;
    for (int k = 0; k < BATCH; ++k) L += bsum[k];
    out[0] = -L;
  }
}

// ---------------------------------------------------------------------------
extern "C" void kernel_launch(void* const* d_in, const int* in_sizes, int n_in,
                              void* d_out, int out_size, void* d_ws, size_t ws_size,
                              hipStream_t stream) {
  (void)in_sizes; (void)n_in; (void)out_size; (void)ws_size;
  const int*   sentence = (const int*)d_in[0];
  const int*   target   = (const int*)d_in[1];
  const unsigned char* mask = (const unsigned char*)d_in[2];
  const float* emb      = (const float*)d_in[3];
  const float* conv_w   = (const float*)d_in[4];
  const float* conv_b   = (const float*)d_in[5];
  const float* lin1_w   = (const float*)d_in[6];
  const float* lin1_b   = (const float*)d_in[7];
  // lstm_params flat: d_in[8 + 4*(2*l + dir) + {Wih,Whh,bih,bhh}]
  const float* h0       = (const float*)d_in[88];
  const float* c0       = (const float*)d_in[89];
  const float* tag_w    = (const float*)d_in[90];
  const float* tag_b    = (const float*)d_in[91];
  const float* crf_start= (const float*)d_in[92];
  const float* crf_end  = (const float*)d_in[93];
  const float* crf_trans= (const float*)d_in[94];

  char* ws = (char*)d_ws;
  size_t off = 0;
  auto alloc = [&](size_t bytes) -> void* {
    void* p = ws + off;
    off = (off + bytes + 255) & ~(size_t)255;
    return p;
  };
  _Float16* Xg   = (_Float16*)alloc((size_t)(SEQ + 2) * BATCH * EMB * 2);
  _Float16* Wg   = (_Float16*)alloc((size_t)3 * 64 * 64 * 64 * 2);
  float* convo   = (float*)alloc((size_t)64 * SEQ * 51 * 4);
  float* pooled  = (float*)alloc((size_t)64 * SEQ * 30 * 4);
  float* x0      = (float*)alloc((size_t)SEQ * BATCH * EMB * 4);
  float* xA      = (float*)alloc((size_t)SEQ * BATCH * 10 * 4);
  float* xB      = (float*)alloc((size_t)SEQ * BATCH * 10 * 4);
  float* preF    = (float*)alloc((size_t)SEQ * BATCH * 20 * 4);
  float* preB    = (float*)alloc((size_t)SEQ * BATCH * 20 * 4);
  float* feats   = (float*)alloc((size_t)BATCH * SEQ * NTAG * 4);

  prep_x_kernel<<<4096, 256, 0, stream>>>(sentence, emb, Xg);
  prep_w_kernel<<<3072, 256, 0, stream>>>(conv_w, Wg);

  const size_t conv_lds = (size_t)(18 * 64 * 114 + 64 * 64) * 2;  // 270848 B
  conv_wmma_kernel<<<SEQ / 16, 512, conv_lds, stream>>>(Xg, Wg, conv_b, convo);

  pool_kernel<<<4096, 256, 0, stream>>>(convo, pooled);
  lin1_kernel<<<SEQ * BATCH, 128, 0, stream>>>(sentence, emb, pooled, lin1_w,
                                               lin1_b, x0);

  const float* xin = x0;
  int in_dim = EMB;
  float* bufs[2] = {xA, xB};
  for (int l = 0; l < 10; ++l) {
    const float* WihF = (const float*)d_in[8 + 4 * (2 * l) + 0];
    const float* WhhF = (const float*)d_in[8 + 4 * (2 * l) + 1];
    const float* bihF = (const float*)d_in[8 + 4 * (2 * l) + 2];
    const float* bhhF = (const float*)d_in[8 + 4 * (2 * l) + 3];
    const float* WihB = (const float*)d_in[8 + 4 * (2 * l + 1) + 0];
    const float* WhhB = (const float*)d_in[8 + 4 * (2 * l + 1) + 1];
    const float* bihB = (const float*)d_in[8 + 4 * (2 * l + 1) + 2];
    const float* bhhB = (const float*)d_in[8 + 4 * (2 * l + 1) + 3];
    const size_t pg_lds = (size_t)(40 * in_dim + 40) * 4;
    lstm_pregate_kernel<<<2048, 256, pg_lds, stream>>>(
        xin, in_dim, WihF, bihF, bhhF, WihB, bihB, bhhB, preF, preB);
    float* xout = bufs[l & 1];
    lstm_scan_kernel<<<2, 256, 0, stream>>>(preF, preB, WhhF, WhhB,
                                            h0 + l * 640, c0 + l * 640, xout);
    xin = xout;
    in_dim = 10;
  }

  feats_kernel<<<5120, 256, 0, stream>>>(xin, tag_w, tag_b, feats);
  crf_kernel<<<1, 640, 0, stream>>>(feats, target, mask, crf_start, crf_end,
                                    crf_trans, (float*)d_out);
}
